// Attention_70403103916776
// MI455X (gfx1250) — matrix operand order repaired and checked by reading, not compiled, not run on hardware
//
#include <hip/hip_runtime.h>
#include <stdint.h>

#define HID 2048
#define NH  16
#define HD  128
#define BB  2
#define TT  2048
#define MTOT (BB*TT)   // 4096 rows of x

typedef __attribute__((ext_vector_type(16))) __bf16   v16bf;
typedef __attribute__((ext_vector_type(8)))  float    v8f;
typedef __attribute__((ext_vector_type(4)))  uint32_t u32x4;

union ABvec {
    u32x4 u[2];   // 32 bytes = 16 bf16 = 8 VGPRs
    v16bf v;
};

__device__ __forceinline__ uint16_t f2bf(float f) {
    uint32_t u = __float_as_uint(f);
    u += 0x7FFFu + ((u >> 16) & 1u);   // round-to-nearest-even
    return (uint16_t)(u >> 16);
}

// ---------------------------------------------------------------- converters
__global__ void cvt_f32_bf16(const float* __restrict__ in,
                             uint16_t* __restrict__ out, int n) {
    int i = blockIdx.x * blockDim.x + threadIdx.x;
    int stride = gridDim.x * blockDim.x;
    for (; i < n; i += stride) out[i] = f2bf(in[i]);
}

// ---------------------------------------------------------------- GEMM
// C[m][n] = sum_k A[m][k] * Bw[n][k]   (i.e. C = A * Bw^T, matching btc,dc->btd)
// A: [M,K] bf16 row-major, Bw: [N,K] bf16 row-major, C: [M,N] fp32.
// Block: 256 threads = 8 waves (2x4); block tile 128x256; wave tile 64x64
// (4x4 WMMA C tiles -> 16 WMMAs per 8 operand-tile loads, intensity 2.0).
__global__ void __launch_bounds__(256)
gemm_bf16_wmma(const uint16_t* __restrict__ A,
               const uint16_t* __restrict__ Bw,
               float* __restrict__ C,
               int M, int N, int K) {
    const int lane = threadIdx.x & 31;
    const int wave = threadIdx.x >> 5;
    const int rowBase = blockIdx.y * 128 + (wave >> 2) * 64;
    const int colBase = blockIdx.x * 256 + (wave & 3) * 64;
    const int lr = lane & 15;
    const int lh = lane >> 4;

    v8f acc[4][4];
#pragma unroll
    for (int i = 0; i < 4; ++i)
#pragma unroll
        for (int j = 0; j < 4; ++j) acc[i][j] = {};

    for (int k = 0; k < K; k += 32) {
        // A operand: lane half lh takes K chunks [k+8*lh, +8) and [k+16+8*lh, +8)
        ABvec a[4];
#pragma unroll
        for (int i = 0; i < 4; ++i) {
            const uint16_t* p = A + (size_t)(rowBase + i*16 + lr) * K + k + lh*8;
            a[i].u[0] = *(const u32x4*)(p);
            a[i].u[1] = *(const u32x4*)(p + 16);
            __builtin_prefetch(p + 128, 0, 3);   // 4 iterations ahead, WGP scope
        }
        // B operand: lane half lh takes K chunks [k+8*lh, +8) and [k+16+8*lh, +8)
        ABvec b[4];
#pragma unroll
        for (int j = 0; j < 4; ++j) {
            const uint16_t* p = Bw + (size_t)(colBase + j*16 + lr) * K + k + lh*8;
            b[j].u[0] = *(const u32x4*)(p);
            b[j].u[1] = *(const u32x4*)(p + 16);
            __builtin_prefetch(p + 128, 0, 3);
        }
#pragma unroll
        for (int i = 0; i < 4; ++i)
#pragma unroll
            for (int j = 0; j < 4; ++j)
                acc[i][j] = __builtin_amdgcn_wmma_f32_16x16x32_bf16(
                    false, a[i].v, false, b[j].v, (short)0, acc[i][j], false, false);
    }

    // C/D layout: lane holds column n = lr; VGPR r holds row m = r + 8*lh
#pragma unroll
    for (int i = 0; i < 4; ++i) {
        const int mB = rowBase + i*16 + lh*8;
#pragma unroll
        for (int j = 0; j < 4; ++j) {
            const int n = colBase + j*16 + lr;
#pragma unroll
            for (int r = 0; r < 8; ++r)
                C[(size_t)(mB + r) * N + n] = acc[i][j][r];
        }
    }
}

// ---------------------------------------------------------------- RoPE + pack
// qf/kf: [B*T, HID] fp32 (row = b*T+t, col = h*128+d)
// qb/kb: [B, H, T, D] bf16, RoPE applied
__global__ void rope_pack(const float* __restrict__ qf, const float* __restrict__ kf,
                          const float* __restrict__ cosT, const float* __restrict__ sinT,
                          uint16_t* __restrict__ qb, uint16_t* __restrict__ kb) {
    int idx = blockIdx.x * blockDim.x + threadIdx.x;      // B*T*NH*64
    if (idx >= BB*TT*NH*(HD/2)) return;
    int i = idx & 63;
    int h = (idx >> 6) & (NH - 1);
    int t = (idx >> 10) & (TT - 1);
    int b = idx >> 21;
    size_t src = (size_t)(b*TT + t) * HID + h*HD + 2*i;
    size_t dst = ((size_t)(b*NH + h) * TT + t) * HD + 2*i;
    float c = cosT[t*64 + i], s = sinT[t*64 + i];
    float xe = qf[src], xo = qf[src + 1];
    qb[dst]     = f2bf(xe*c - xo*s);
    qb[dst + 1] = f2bf(xe*s + xo*c);
    xe = kf[src]; xo = kf[src + 1];
    kb[dst]     = f2bf(xe*c - xo*s);
    kb[dst + 1] = f2bf(xe*s + xo*c);
}

// ---------------------------------------------------------------- V transpose
// vf: [B*T, HID] fp32  ->  vt: [B, H, D, T] bf16
__global__ void v_transpose(const float* __restrict__ vf, uint16_t* __restrict__ vt) {
    int idx = blockIdx.x * blockDim.x + threadIdx.x;      // B*NH*HD*TT
    if (idx >= BB*NH*HD*TT) return;
    int t  = idx & (TT - 1);
    int d  = (idx >> 11) & (HD - 1);
    int bh = idx >> 18;
    int b = bh >> 4, h = bh & 15;
    vt[idx] = f2bf(vf[(size_t)(b*TT + t) * HID + h*HD + d]);
}

// ---------------------------------------------------------------- flash attention
// One wave handles a 16-query tile of one (b,h). Iterates 32 keys at a time.
__global__ void __launch_bounds__(32)
flash_attn(const uint16_t* __restrict__ qb, const uint16_t* __restrict__ kb,
           const uint16_t* __restrict__ vt, uint16_t* __restrict__ ob) {
    __shared__ uint16_t ldsP[16 * 32];     // P tile bounce: C-layout -> A-layout
    const int lane = threadIdx.x & 31;
    const int lr = lane & 15, lh = lane >> 4;
    const int q0 = blockIdx.x * 16;
    const int bh = blockIdx.y;
    const int b = bh >> 4, h = bh & 15;

    const uint16_t* Qh = qb + (size_t)bh * TT * HD;
    const uint16_t* Kh = kb + (size_t)bh * TT * HD;
    const uint16_t* Vh = vt + (size_t)bh * HD * TT;

    // Q as 4 A-matrices (head dim K chunks of 32), held in registers for whole pass
    ABvec aQ[4];
#pragma unroll
    for (int c = 0; c < 4; ++c) {
        const uint16_t* p = Qh + (size_t)(q0 + lr) * HD + c*32 + lh*8;
        aQ[c].u[0] = *(const u32x4*)(p);
        aQ[c].u[1] = *(const u32x4*)(p + 16);
    }

    v8f o[8];
#pragma unroll
    for (int jd = 0; jd < 8; ++jd) o[jd] = {};
    float mrun[8], lrun[8];
#pragma unroll
    for (int r = 0; r < 8; ++r) { mrun[r] = -1e30f; lrun[r] = 0.0f; }

    const float scale = 0.08838834764831845f;   // 1/sqrt(128)
    const int kend = q0 + 16;
    for (int kb0 = 0; kb0 < kend; kb0 += 32) {
        // ---- S = Q K^T for 32 keys (two 16-key C tiles)
        v8f s0 = {}, s1 = {};
#pragma unroll
        for (int c = 0; c < 4; ++c) {
            ABvec bK0, bK1;
            const uint16_t* p0 = Kh + (size_t)(kb0 + lr) * HD + c*32 + lh*8;
            bK0.u[0] = *(const u32x4*)(p0);
            bK0.u[1] = *(const u32x4*)(p0 + 16);
            const uint16_t* p1 = Kh + (size_t)(kb0 + 16 + lr) * HD + c*32 + lh*8;
            bK1.u[0] = *(const u32x4*)(p1);
            bK1.u[1] = *(const u32x4*)(p1 + 16);
            s0 = __builtin_amdgcn_wmma_f32_16x16x32_bf16(
                false, aQ[c].v, false, bK0.v, (short)0, s0, false, false);
            s1 = __builtin_amdgcn_wmma_f32_16x16x32_bf16(
                false, aQ[c].v, false, bK1.v, (short)0, s1, false, false);
        }

        // prefetch next key block's K and V while we do the softmax VALU work
        if (kb0 + 32 < kend) {
            __builtin_prefetch(Kh + (size_t)(kb0 + 32 + lr) * HD + lh*16, 0, 3);
            __builtin_prefetch(Vh + (size_t)(lr) * TT + kb0 + 32 + lh*16, 0, 3);
        }

        // ---- causal mask + online softmax (rows live in 16-lane halves)
        float p0a[8], p1a[8], alph[8];
#pragma unroll
        for (int r = 0; r < 8; ++r) {
            int qq = q0 + r + lh*8;
            float v0 = s0[r] * scale;
            float v1 = s1[r] * scale;
            if (kb0 + lr > qq)      v0 = -1e30f;
            if (kb0 + 16 + lr > qq) v1 = -1e30f;
            float t = fmaxf(v0, v1);
            t = fmaxf(t, __shfl_xor(t, 1, 32));
            t = fmaxf(t, __shfl_xor(t, 2, 32));
            t = fmaxf(t, __shfl_xor(t, 4, 32));
            t = fmaxf(t, __shfl_xor(t, 8, 32));
            float mnew = fmaxf(mrun[r], t);
            float a  = __expf(mrun[r] - mnew);
            float p0 = __expf(v0 - mnew);
            float p1 = __expf(v1 - mnew);
            float sm = p0 + p1;
            sm += __shfl_xor(sm, 1, 32);
            sm += __shfl_xor(sm, 2, 32);
            sm += __shfl_xor(sm, 4, 32);
            sm += __shfl_xor(sm, 8, 32);
            lrun[r] = lrun[r] * a + sm;
            mrun[r] = mnew;
            alph[r] = a;
            p0a[r] = p0; p1a[r] = p1;
        }
#pragma unroll
        for (int jd = 0; jd < 8; ++jd)
#pragma unroll
            for (int r = 0; r < 8; ++r) o[jd][r] *= alph[r];

        // ---- reshape P (C layout) -> A layout via LDS
#pragma unroll
        for (int r = 0; r < 8; ++r) {
            int m = r + lh*8;
            ldsP[m*32 + lr]      = f2bf(p0a[r]);
            ldsP[m*32 + 16 + lr] = f2bf(p1a[r]);
        }
        __syncthreads();
        ABvec aP;
        {
            const uint16_t* p = &ldsP[lr*32 + lh*8];
            aP.u[0] = *(const u32x4*)(p);
            aP.u[1] = *(const u32x4*)(p + 16);
        }
        __syncthreads();

        // ---- O += P V  (V^T layout makes B-operand loads contiguous)
#pragma unroll
        for (int jd = 0; jd < 8; ++jd) {
            ABvec bV;
            const uint16_t* p = Vh + (size_t)(jd*16 + lr) * TT + kb0 + lh*8;
            bV.u[0] = *(const u32x4*)(p);
            bV.u[1] = *(const u32x4*)(p + 16);
            o[jd] = __builtin_amdgcn_wmma_f32_16x16x32_bf16(
                false, aP.v, false, bV.v, (short)0, o[jd], false, false);
        }
    }

    // ---- normalize and store to [B*T, HID] bf16 for the output projection
#pragma unroll
    for (int r = 0; r < 8; ++r) {
        float inv = 1.0f / lrun[r];
        int m = r + lh*8;
        size_t row = (size_t)(b*TT + q0 + m) * HID + (size_t)h * HD;
#pragma unroll
        for (int jd = 0; jd < 8; ++jd)
            ob[row + jd*16 + lr] = f2bf(o[jd][r] * inv);
    }
}

// ---------------------------------------------------------------- launcher
extern "C" void kernel_launch(void* const* d_in, const int* in_sizes, int n_in,
                              void* d_out, int out_size, void* d_ws, size_t ws_size,
                              hipStream_t stream) {
    (void)in_sizes; (void)n_in; (void)out_size; (void)ws_size;
    const float* x  = (const float*)d_in[0];
    const float* wq = (const float*)d_in[1];
    const float* wk = (const float*)d_in[2];
    const float* wv = (const float*)d_in[3];
    const float* wo = (const float*)d_in[4];
    const float* fc = (const float*)d_in[5];
    const float* fs = (const float*)d_in[6];
    // d_in[7] = mask, implied by causal structure

    char* ws = (char*)d_ws;
    size_t off = 0;
    auto alloc = [&](size_t bytes) -> char* {
        char* p = ws + off;
        off += (bytes + 255) & ~(size_t)255;
        return p;
    };
    uint16_t* xb  = (uint16_t*)alloc((size_t)MTOT * HID * 2);
    uint16_t* wqb = (uint16_t*)alloc((size_t)HID * HID * 2);
    uint16_t* wkb = (uint16_t*)alloc((size_t)HID * HID * 2);
    uint16_t* wvb = (uint16_t*)alloc((size_t)HID * HID * 2);
    uint16_t* wob = (uint16_t*)alloc((size_t)HID * HID * 2);
    float*    qf  = (float*)   alloc((size_t)MTOT * HID * 4);
    float*    kf  = (float*)   alloc((size_t)MTOT * HID * 4);
    float*    vf  = (float*)   alloc((size_t)MTOT * HID * 4);
    uint16_t* qbb = (uint16_t*)alloc((size_t)BB * NH * TT * HD * 2);
    uint16_t* kbb = (uint16_t*)alloc((size_t)BB * NH * TT * HD * 2);
    uint16_t* vtb = (uint16_t*)alloc((size_t)BB * NH * HD * TT * 2);
    uint16_t* ob  = (uint16_t*)qf;   // qf is dead after rope_pack; reuse for attn output

    // 1) fp32 -> bf16
    cvt_f32_bf16<<<2048, 256, 0, stream>>>(x,  xb,  MTOT * HID);
    cvt_f32_bf16<<<1024, 256, 0, stream>>>(wq, wqb, HID * HID);
    cvt_f32_bf16<<<1024, 256, 0, stream>>>(wk, wkb, HID * HID);
    cvt_f32_bf16<<<1024, 256, 0, stream>>>(wv, wvb, HID * HID);
    cvt_f32_bf16<<<1024, 256, 0, stream>>>(wo, wob, HID * HID);

    // 2) QKV projections (C = X * W^T), fp32 out
    dim3 gg(HID / 256, MTOT / 128);
    gemm_bf16_wmma<<<gg, 256, 0, stream>>>(xb, wqb, qf, MTOT, HID, HID);
    gemm_bf16_wmma<<<gg, 256, 0, stream>>>(xb, wkb, kf, MTOT, HID, HID);
    gemm_bf16_wmma<<<gg, 256, 0, stream>>>(xb, wvb, vf, MTOT, HID, HID);

    // 3) RoPE + pack bf16, 4) V transpose
    rope_pack<<<(BB*TT*NH*(HD/2) + 255) / 256, 256, 0, stream>>>(qf, kf, fc, fs, qbb, kbb);
    v_transpose<<<(BB*NH*HD*TT + 255) / 256, 256, 0, stream>>>(vf, vtb);

    // 5) flash attention (one wave per 16-query tile per head)
    flash_attn<<<dim3(TT / 16, BB * NH), 32, 0, stream>>>(qbb, kbb, vtb, ob);

    // 6) output projection -> d_out fp32
    gemm_bf16_wmma<<<gg, 256, 0, stream>>>(ob, wob, (float*)d_out, MTOT, HID, HID);
}
